// MyMultiheadAttention_82317343195899
// MI455X (gfx1250) — matrix operand order repaired and checked by reading, not compiled, not run on hardware
//
#include <hip/hip_runtime.h>

typedef __attribute__((ext_vector_type(16))) __bf16 v16bf;
typedef __attribute__((ext_vector_type(8)))  __bf16 v8bf;
typedef __attribute__((ext_vector_type(8)))  float  v8f;

#define EMB   192
#define NH    3
#define HD    64
#define NTOK  1024
#define NBAT  32
#define SM_SCALE 0.125f  /* 64^-0.5 */

// Padded LDS row strides (keep 32-byte alignment, break 64-bank periodicity)
#define S_STRIDE 1028   /* floats */
#define P_STRIDE 1040   /* bf16   */

static __device__ __forceinline__ v8f wmma_bf16(v16bf a, v16bf b, v8f c) {
    return __builtin_amdgcn_wmma_f32_16x16x32_bf16(false, a, false, b, (short)0, c, false, false);
}

static __device__ __forceinline__ v16bf pack_a_bf16(const __bf16* base, int off_lo, int off_hi) {
    v8bf f0 = *(const v8bf*)(base + off_lo);
    v8bf f1 = *(const v8bf*)(base + off_hi);
    v16bf a;
#pragma unroll
    for (int i = 0; i < 8; ++i) { a[i] = f0[i]; a[i + 8] = f1[i]; }
    return a;
}

// WMMA A-fragment element order within a 32-key block: key l -> slot perm5(l).
// perm5(K_local(e,hi)) == hi*16 + e, so a fragment is one contiguous 32-B read.
static __device__ __forceinline__ int perm5(int l) {
    return (l & 7) | ((l & 8) << 1) | ((l & 16) >> 1);
}

// ---------------------------------------------------------------------------
// Kernel 0: convert + transpose weights to bf16 so B-tiles are contiguous.
// ---------------------------------------------------------------------------
__global__ __launch_bounds__(256) void prep_kernel(
    const float* __restrict__ w_qkv, const float* __restrict__ w_out,
    __bf16* __restrict__ wqkvT, __bf16* __restrict__ woutT)
{
    int i = blockIdx.x * 256 + threadIdx.x;
    if (i < 576 * 192) {
        int c = i / 192, k = i % 192;
        wqkvT[i] = (__bf16)w_qkv[k * 576 + c];
    } else {
        int j = i - 576 * 192;
        if (j < 192 * 192) {
            int c = j / 192, k = j % 192;
            woutT[j] = (__bf16)w_out[k * 192 + c];
        }
    }
}

// ---------------------------------------------------------------------------
// Kernel 1: QKV projection.  One wave = one 16-token M-tile; A (16x192) is
// register-resident.  N-tiles paired for dual independent WMMA chains.
// ---------------------------------------------------------------------------
__global__ __launch_bounds__(128) void qkv_kernel(
    const float* __restrict__ x, const float* __restrict__ b_qkv,
    const __bf16* __restrict__ wT,
    __bf16* __restrict__ q_ws, __bf16* __restrict__ k_ws, __bf16* __restrict__ v_ws)
{
    const int lane = threadIdx.x & 31;
    const int wave = threadIdx.x >> 5;
    const int hi = lane >> 4, lo = lane & 15;
    const int tok0 = (blockIdx.x * 4 + wave) * 16;
    const float* xr = x + (size_t)(tok0 + lo) * EMB;

    v16bf A[6];
#pragma unroll
    for (int ks = 0; ks < 6; ++ks) {
        v8f f0 = *(const v8f*)(xr + ks * 32 + hi * 16);
        v8f f1 = *(const v8f*)(xr + ks * 32 + hi * 16 + 8);
        v16bf a;
#pragma unroll
        for (int i = 0; i < 8; ++i) { a[i] = (__bf16)f0[i]; a[i + 8] = (__bf16)f1[i]; }
        A[ks] = a;
    }

    const int tokb = tok0 + hi * 8;
    const int bb = tokb >> 10;
    const int nb = tokb & 1023;

    // ---- Q (c in [0,192)) and K (c in [192,384)) -------------------------
#pragma unroll 1
    for (int grp = 0; grp < 2; ++grp) {
        __bf16* __restrict__ dst = grp ? k_ws : q_ws;
        const int cbase = grp * 192;
        for (int nt = 0; nt < 12; nt += 2) {
            const int c0 = cbase + nt * 16 + lo;
            const int c1 = c0 + 16;
            const __bf16* w0 = wT + (size_t)c0 * EMB;
            const __bf16* w1 = wT + (size_t)c1 * EMB;
            v8f acc0 = {}, acc1 = {};
#pragma unroll
            for (int ks = 0; ks < 6; ++ks) {
                v16bf b0 = *(const v16bf*)(w0 + ks * 32 + hi * 16);
                v16bf b1 = *(const v16bf*)(w1 + ks * 32 + hi * 16);
                acc0 = wmma_bf16(A[ks], b0, acc0);
                acc1 = wmma_bf16(A[ks], b1, acc1);
            }
            const float bias0 = b_qkv[c0], bias1 = b_qkv[c1];
            const int h0 = (nt * 16) / 64,  h1 = ((nt + 1) * 16) / 64;
            const int d0 = (nt * 16) % 64 + lo, d1 = ((nt + 1) * 16) % 64 + lo;
            __bf16* o0 = dst + ((size_t)((bb * NH + h0) * NTOK + nb)) * HD + d0;
            __bf16* o1 = dst + ((size_t)((bb * NH + h1) * NTOK + nb)) * HD + d1;
#pragma unroll
            for (int r = 0; r < 8; ++r) {
                o0[(size_t)r * HD] = (__bf16)(acc0[r] + bias0);
                o1[(size_t)r * HD] = (__bf16)(acc1[r] + bias1);
            }
        }
    }

    // ---- V (c in [384,576)), transposed + packed 16-B stores -------------
    for (int nt = 0; nt < 12; nt += 2) {
        const int c0 = 384 + nt * 16 + lo;
        const int c1 = c0 + 16;
        const __bf16* w0 = wT + (size_t)c0 * EMB;
        const __bf16* w1 = wT + (size_t)c1 * EMB;
        v8f acc0 = {}, acc1 = {};
#pragma unroll
        for (int ks = 0; ks < 6; ++ks) {
            v16bf b0 = *(const v16bf*)(w0 + ks * 32 + hi * 16);
            v16bf b1 = *(const v16bf*)(w1 + ks * 32 + hi * 16);
            acc0 = wmma_bf16(A[ks], b0, acc0);
            acc1 = wmma_bf16(A[ks], b1, acc1);
        }
        const float bias0 = b_qkv[c0], bias1 = b_qkv[c1];
        const int h0 = (nt * 16) / 64,  h1 = ((nt + 1) * 16) / 64;
        const int d0 = (nt * 16) % 64 + lo, d1 = ((nt + 1) * 16) % 64 + lo;
        v8bf pk0, pk1;
#pragma unroll
        for (int r = 0; r < 8; ++r) {
            pk0[r] = (__bf16)(acc0[r] + bias0);
            pk1[r] = (__bf16)(acc1[r] + bias1);
        }
        *(v8bf*)(v_ws + ((size_t)((bb * NH + h0) * HD + d0)) * NTOK + nb) = pk0;
        *(v8bf*)(v_ws + ((size_t)((bb * NH + h1) * HD + d1)) * NTOK + nb) = pk1;
    }
}

// ---------------------------------------------------------------------------
// Kernel 2: attention.  One workgroup (4 waves) = one (p, 16-query block).
// Full 16x1024 score strip + natural-order bf16 P strip in LDS (~98 KB of
// 320 KB WGP LDS).  Padded strides kill the 64-bank periodicity.
// ---------------------------------------------------------------------------
__global__ __launch_bounds__(128) void attn_kernel(
    const __bf16* __restrict__ q_ws, const __bf16* __restrict__ k_ws,
    const __bf16* __restrict__ v_ws, __bf16* __restrict__ attn)
{
    __shared__ float S[16][S_STRIDE];
    __shared__ __attribute__((aligned(32))) __bf16 P[16][P_STRIDE];
    __shared__ float red[16][8];
    __shared__ float rcp[16];

    const int lane = threadIdx.x & 31;
    const int wave = threadIdx.x >> 5;
    const int hi = lane >> 4, lo = lane & 15;
    const int p  = blockIdx.y;          // b*3 + h
    const int n0 = blockIdx.x * 16;

    // --- Phase 1: S = (Q Kt) * scale, dual-tile chains --------------------
    const __bf16* qrow = q_ws + (size_t)(p * NTOK + n0 + lo) * HD;
    v16bf AQ[2];
#pragma unroll
    for (int ks = 0; ks < 2; ++ks)
        AQ[ks] = pack_a_bf16(qrow, ks * 32 + hi * 16, ks * 32 + hi * 16 + 8);

    for (int it = 0; it < 8; ++it) {
        const int t0 = wave + it * 8, t1 = t0 + 4;
        const __bf16* k0 = k_ws + (size_t)(p * NTOK + t0 * 16 + lo) * HD;
        const __bf16* k1 = k_ws + (size_t)(p * NTOK + t1 * 16 + lo) * HD;
        v8f acc0 = {}, acc1 = {};
#pragma unroll
        for (int ks = 0; ks < 2; ++ks) {
            v16bf b0 = *(const v16bf*)(k0 + ks * 32 + hi * 16);
            v16bf b1 = *(const v16bf*)(k1 + ks * 32 + hi * 16);
            acc0 = wmma_bf16(AQ[ks], b0, acc0);
            acc1 = wmma_bf16(AQ[ks], b1, acc1);
        }
#pragma unroll
        for (int r = 0; r < 8; ++r) {
            S[hi * 8 + r][t0 * 16 + lo] = acc0[r] * SM_SCALE;
            S[hi * 8 + r][t1 * 16 + lo] = acc1[r] * SM_SCALE;
        }
    }
    __syncthreads();

    // --- Phase 2: row softmax; lanes span all 16 rows (row = t&15) --------
    {
        const int row = threadIdx.x & 15, seg = threadIdx.x >> 4;
        const int c0 = seg * 128;
        float m = -3.0e38f;
        for (int c = 0; c < 128; ++c) m = fmaxf(m, S[row][c0 + c]);
        red[row][seg] = m;
        __syncthreads();
        float rm = red[row][0];
#pragma unroll
        for (int i = 1; i < 8; ++i) rm = fmaxf(rm, red[row][i]);
        __syncthreads();
        float s = 0.f;
        for (int c = 0; c < 128; ++c) {
            const int cabs = c0 + c;
            float e = __expf(S[row][cabs] - rm);
            // store in natural key order (contiguous A-fragment order k = 16h+s)
            P[row][cabs] = (__bf16)e;
            s += e;
        }
        red[row][seg] = s;
        __syncthreads();
        if (seg == 0) {
            float tot = 0.f;
#pragma unroll
            for (int i = 0; i < 8; ++i) tot += red[row][i];
            rcp[row] = 1.0f / tot;
        }
        __syncthreads();
    }

    // --- Phase 3: O = P @ V; A fragments are single contiguous LDS reads --
    const int d = wave * 16 + lo;
    const __bf16* vrow = v_ws + (size_t)(p * HD + d) * NTOK;
    v8f acc0 = {}, acc1 = {};
    for (int kk = 0; kk < 1024; kk += 64) {
        v16bf a0 = *(const v16bf*)&P[lo][kk + hi * 16];
        v16bf a1 = *(const v16bf*)&P[lo][kk + 32 + hi * 16];
        v16bf b0 = *(const v16bf*)(vrow + kk + hi * 16);
        v16bf b1 = *(const v16bf*)(vrow + kk + 32 + hi * 16);
        acc0 = wmma_bf16(a0, b0, acc0);
        acc1 = wmma_bf16(a1, b1, acc1);
    }
    const int bb = p / NH, hh = p % NH;
    __bf16* orow = attn + ((size_t)bb * NTOK + n0 + hi * 8) * EMB + hh * HD + d;
#pragma unroll
    for (int r = 0; r < 8; ++r) {
        const float v = (acc0[r] + acc1[r]) * rcp[hi * 8 + r];
        orow[(size_t)r * EMB] = (__bf16)v;
    }
}

// ---------------------------------------------------------------------------
// Kernel 3: output projection.  out = attn @ w_out + b_out (fp32 out).
// ---------------------------------------------------------------------------
__global__ __launch_bounds__(128) void proj_kernel(
    const __bf16* __restrict__ attn, const __bf16* __restrict__ wT,
    const float* __restrict__ b_out, float* __restrict__ out)
{
    const int lane = threadIdx.x & 31;
    const int wave = threadIdx.x >> 5;
    const int hi = lane >> 4, lo = lane & 15;
    const int tok0 = (blockIdx.x * 4 + wave) * 16;
    const __bf16* arow = attn + (size_t)(tok0 + lo) * EMB;

    v16bf A[6];
#pragma unroll
    for (int ks = 0; ks < 6; ++ks)
        A[ks] = pack_a_bf16(arow, ks * 32 + hi * 16, ks * 32 + hi * 16 + 8);

    for (int nt = 0; nt < 12; nt += 2) {
        const int c0 = nt * 16 + lo, c1 = c0 + 16;
        const __bf16* w0 = wT + (size_t)c0 * EMB;
        const __bf16* w1 = wT + (size_t)c1 * EMB;
        v8f acc0 = {}, acc1 = {};
#pragma unroll
        for (int ks = 0; ks < 6; ++ks) {
            v16bf b0 = *(const v16bf*)(w0 + ks * 32 + hi * 16);
            v16bf b1 = *(const v16bf*)(w1 + ks * 32 + hi * 16);
            acc0 = wmma_bf16(A[ks], b0, acc0);
            acc1 = wmma_bf16(A[ks], b1, acc1);
        }
        const float bias0 = b_out[c0], bias1 = b_out[c1];
        float* o = out + (size_t)(tok0 + hi * 8) * EMB;
#pragma unroll
        for (int r = 0; r < 8; ++r) {
            o[(size_t)r * EMB + c0] = acc0[r] + bias0;
            o[(size_t)r * EMB + c1] = acc1[r] + bias1;
        }
    }
}

// ---------------------------------------------------------------------------
extern "C" void kernel_launch(void* const* d_in, const int* in_sizes, int n_in,
                              void* d_out, int out_size, void* d_ws, size_t ws_size,
                              hipStream_t stream)
{
    const float* x     = (const float*)d_in[0];
    const float* w_qkv = (const float*)d_in[1];
    const float* b_qkv = (const float*)d_in[2];
    const float* w_out = (const float*)d_in[3];
    const float* b_out = (const float*)d_in[4];
    float* out = (float*)d_out;

    const size_t SZ_QKV = (size_t)NBAT * NH * NTOK * HD * sizeof(__bf16); // 12 MB each
    char* ws = (char*)d_ws;
    __bf16* q_ws  = (__bf16*)(ws);
    __bf16* k_ws  = (__bf16*)(ws + SZ_QKV);
    __bf16* v_ws  = (__bf16*)(ws + 2 * SZ_QKV);
    __bf16* attn  = (__bf16*)(ws + 3 * SZ_QKV);                 // [32768][192] bf16
    __bf16* wqkvT = (__bf16*)(ws + 4 * SZ_QKV);
    __bf16* woutT = (__bf16*)(ws + 4 * SZ_QKV + 576 * 192 * sizeof(__bf16));

    prep_kernel<<<576, 256, 0, stream>>>(w_qkv, w_out, wqkvT, woutT);
    qkv_kernel<<<512, 128, 0, stream>>>(x, b_qkv, wqkvT, q_ws, k_ws, v_ws);
    attn_kernel<<<dim3(64, 96), 128, 0, stream>>>(q_ws, k_ws, v_ws, attn);
    proj_kernel<<<512, 128, 0, stream>>>(attn, woutT, b_out, out);
}
